// PIGNODE_44779329028552
// MI455X (gfx1250) — matrix operand order repaired and checked by reading, not compiled, not run on hardware
//
#include <hip/hip_runtime.h>
#include <math.h>

// ---------------- problem constants ----------------
#define GG      64
#define NNODE   4096                // 64*64
#define BATCH   4
#define MTOT    (BATCH*NNODE)       // 16384 nodes total
#define HID     64
#define HEADS   4
#define NCOL    (HEADS*HID)         // 256
#define LDA     72                  // padded LDS row stride (ushorts) = 144B

typedef __attribute__((ext_vector_type(16))) __bf16 v16bf;
typedef __attribute__((ext_vector_type(8)))  float  v8f;
typedef __attribute__((ext_vector_type(4)))  unsigned u32x4;
typedef __attribute__((ext_vector_type(4)))  int      i32x4;
typedef __attribute__((ext_vector_type(8)))  int      i32x8;

#if __has_builtin(__builtin_amdgcn_tensor_load_to_lds) && __has_builtin(__builtin_amdgcn_s_wait_tensorcnt)
#define USE_TDM 1
#else
#define USE_TDM 0
#endif
#if __has_include(<hip/amd_detail/amd_gfx1250_TDM.h>)
#define TDM_SIX_ARG 1
#else
#define TDM_SIX_ARG 0
#endif

union BFrag { v16bf v; unsigned u[8]; };

__device__ __forceinline__ unsigned short f2bf(float f) {
  unsigned u = __builtin_bit_cast(unsigned, f);
  u += 0x7FFFu + ((u >> 16) & 1u);          // round-to-nearest-even
  return (unsigned short)(u >> 16);
}
__device__ __forceinline__ unsigned packbf(float a, float b) {
#if __has_builtin(__builtin_amdgcn_cvt_pk_bf16_f32)
  auto r = __builtin_amdgcn_cvt_pk_bf16_f32(a, b);   // v_cvt_pk_bf16_f32
  return __builtin_bit_cast(unsigned, r);
#else
  // native __bf16 casts: let ISel pick v_cvt_bf16_f32 / packed form on gfx1250
  union { __bf16 h[2]; unsigned u; } r;
  r.h[0] = (__bf16)a; r.h[1] = (__bf16)b;
  return r.u;
#endif
}
__device__ __forceinline__ float bf2f(unsigned s16) {
  return __builtin_bit_cast(float, s16 << 16);
}
__device__ __forceinline__ float silu_f(float v) { return v / (1.f + __expf(-v)); }

// neighbor directions in reference order: dy in {-1,0,1}, dx in {-1,0,1}, skip (0,0)
__constant__ int   c_dx[8]  = {-1, 0, 1,-1, 1,-1, 0, 1};
__constant__ int   c_dy[8]  = {-1,-1,-1, 0, 0, 1, 1, 1};
__constant__ float c_ndx[8] = {-0.70710678f, 0.f, 0.70710678f,-1.f, 1.f,-0.70710678f, 0.f, 0.70710678f};
__constant__ float c_ndy[8] = {-0.70710678f,-1.f,-0.70710678f, 0.f, 0.f, 0.70710678f, 1.f, 0.70710678f};

// ---------------- weight pre-conversion: wbf[L][n][k] = bf16(W_L[k][n]) ----------------
__global__ void convw_k(const float* __restrict__ W0, const float* __restrict__ W1,
                        unsigned short* __restrict__ wbf) {
  int i = blockIdx.x * blockDim.x + threadIdx.x;   // 2 * 256 * 64
  int L = i >> 14, r = i & 16383;
  int n = r >> 6, k = r & 63;
  const float* W = L ? W1 : W0;
  wbf[(size_t)L * NCOL * HID + n * HID + k] = f2bf(W[(size_t)k * NCOL + n]);
}

// ---- bf16 WMMA GEMM fused with attention-score epilogue -------------------------------
// xh(M x 256) = A(M x 64) @ W(64 x 256); a_src/a_dst[node][head] = dot(xh row, att vec).
// Block = 256 threads (8 waves), tile 128 rows. Weights staged into padded LDS by the
// Tensor Data Mover (pad 4 DW every 32 DW -> 144B rows), overlapped with A conversion.
__global__ void wmma_gemm64(const float* __restrict__ A,
                            const unsigned short* __restrict__ Wbf,   // [256][64] bf16
                            const float* __restrict__ att_src,        // [256]
                            const float* __restrict__ att_dst,        // [256]
                            unsigned short* __restrict__ Out,
                            float* __restrict__ a_src, float* __restrict__ a_dst) {
  __shared__ __align__(16) unsigned short As[128 * LDA];   // 18 KB (reused as D-tile stage)
  __shared__ __align__(16) unsigned short Wt[NCOL * LDA];  // 36 KB padded [n][k]
  const int tid = threadIdx.x;
  const int blockRow = blockIdx.x * 128;

#if USE_TDM
  if (tid < 32) {                     // wave 0 issues one TDM descriptor for the 32KB weight tile
    u32x4 g0; i32x8 g1;
    i32x4 g2 = {0, 0, 0, 0}, g3 = {0, 0, 0, 0};
    unsigned long long ga = (unsigned long long)Wbf;
    g0[0] = 1u;                                               // count=1 (valid descriptor)
    g0[1] = (unsigned)(unsigned long long)(const void*)Wt;    // lds_addr (low 32 = LDS offset)
    g0[2] = (unsigned)ga;                                     // global_addr[31:0]
    g0[3] = (unsigned)((ga >> 32) & 0x1FFFFFFull) | 0x80000000u; // addr[56:32] | type=2
    const unsigned NEL = NCOL * HID;                          // 16384 bf16 elements
    g1[0] = (int)((1u << 16) | (1u << 20) | (4u << 22) | (3u << 25)); // 2B elems, pad 4DW/32DW
    g1[1] = (int)((NEL & 0xFFFFu) << 16);                     // tensor_dim0 lo16
    g1[2] = (int)((NEL >> 16) | (1u << 16));                  // tensor_dim0 hi16, tensor_dim1=1
    g1[3] = (int)((NEL & 0xFFFFu) << 16);                     // tile_dim0 = 16384
    g1[4] = 1;                                                // tile_dim1 = 1
    g1[5] = (int)NEL;                                         // tensor_dim0_stride lo32
    g1[6] = 0; g1[7] = 0;
#if TDM_SIX_ARG
    i32x8 g4 = {0, 0, 0, 0, 0, 0, 0, 0};
    __builtin_amdgcn_tensor_load_to_lds(g0, g1, g2, g3, g4, 0);
#else
    __builtin_amdgcn_tensor_load_to_lds(g0, g1, g2, g3, 0);
#endif
  }
#endif

  // stage A (f32 -> packed bf16 pairs), padded rows; overlaps the TDM weight DMA
  {
    const float2* A2 = (const float2*)A;
    for (int i = tid; i < 128 * 32; i += 256) {
      int r = i >> 5, cp = i & 31;
      float2 v = A2[(size_t)(blockRow + r) * 32 + cp];
      *(unsigned*)&As[r * LDA + 2 * cp] = packbf(v.x, v.y);
    }
  }

#if USE_TDM
  if (tid < 32) __builtin_amdgcn_s_wait_tensorcnt(0);
#else
  for (int i = tid; i < NCOL * 8; i += 256) {                 // manual padded 16B copy fallback
    int row = i >> 3, seg = i & 7;
    *(uint4*)&Wt[row * LDA + seg * 8] = ((const uint4*)Wbf)[i];
  }
#endif
  __syncthreads();

  const int wave = tid >> 5;
  const int lane = tid & 31;
  const int half = lane >> 4;
  const int m16  = lane & 15;
  const int rowBase = wave * 16;

  // A fragments for both K-steps (ISA 16-bit A layout: half-wave splits K at 8)
  BFrag afr[2];
#pragma unroll
  for (int kk = 0; kk < 2; ++kk)
#pragma unroll
    for (int v = 0; v < 8; ++v) {
      int k0 = kk * 32 + (v < 4 ? 2 * v + 8 * half : 16 + 2 * (v - 4) + 8 * half);
      afr[kk].u[v] = *(const unsigned*)&As[(rowBase + m16) * LDA + k0];
    }

  // each wave's own As rows are now dead -> reuse as 16x16 f32 D-tile staging area
  float* ot = (float*)(As + wave * 16 * LDA);
  const int r2  = lane >> 1;           // output row within tile for packed store
  const int seg = (lane & 1) * 8;      // output col segment (8 bf16 = 16B)
  float asum[HEADS] = {0.f, 0.f, 0.f, 0.f}, dsum[HEADS] = {0.f, 0.f, 0.f, 0.f};

  // outer loop over heads fully unrolled (hq compile-time -> fixed accumulators);
  // inner column-tile loop kept mostly rolled to bound register pressure (no spills).
#pragma unroll
  for (int hq = 0; hq < HEADS; ++hq) {
#pragma unroll 2
    for (int t = 0; t < 4; ++t) {
      const int nt = hq * 4 + t;
      v8f acc = {0.f, 0.f, 0.f, 0.f, 0.f, 0.f, 0.f, 0.f};
#pragma unroll
      for (int kk = 0; kk < 2; ++kk) {
        BFrag bfr;
#pragma unroll
        for (int v = 0; v < 8; ++v) {
          int k0 = kk * 32 + (v < 4 ? 2 * v + 8 * half : 16 + 2 * (v - 4) + 8 * half);    // B layout: half-wave splits K at 8 (same as A)
          bfr.u[v] = *(const unsigned*)&Wt[(nt * 16 + m16) * LDA + k0];
        }
        acc = __builtin_amdgcn_wmma_f32_16x16x32_bf16(false, afr[kk].v,
                                                      false, bfr.v,
                                                      (short)0, acc, false, false);
      }
      // C/D layout: VGPR r -> row = r + 8*half, col = lane&15. Stage via LDS, re-read
      // row-major, fold in attention-score partials, then one global_store_b128.
#pragma unroll
      for (int r = 0; r < 8; ++r) ot[(r + 8 * half) * 16 + m16] = acc[r];
      unsigned pk[4];
      float pa = 0.f, pd = 0.f;
#pragma unroll
      for (int j = 0; j < 4; ++j) {
        float a = ot[r2 * 16 + seg + 2 * j];
        float b = ot[r2 * 16 + seg + 2 * j + 1];
        int gc = nt * 16 + seg + 2 * j;  // global column == h*64 + c
        pa += a * att_src[gc] + b * att_src[gc + 1];
        pd += a * att_dst[gc] + b * att_dst[gc + 1];
        pk[j] = packbf(a, b);
      }
      asum[hq] += pa; dsum[hq] += pd;
      uint4 st = {pk[0], pk[1], pk[2], pk[3]};
      *(uint4*)&Out[(size_t)(blockRow + rowBase + r2) * NCOL + nt * 16 + seg] = st;
    }
  }

  // finalize a_src/a_dst: combine the two 8-col halves of each row (lane pairs)
#pragma unroll
  for (int h = 0; h < HEADS; ++h) {
    float va = asum[h] + __shfl_xor(asum[h], 1, 32);
    float vd = dsum[h] + __shfl_xor(dsum[h], 1, 32);
    if ((lane & 1) == 0) {
      int row = blockRow + rowBase + r2;
      a_src[row * 4 + h] = va;
      a_dst[row * 4 + h] = vd;
    }
  }
}

// ---------------- fused softmax-aggregate + mean(heads)+bias + LayerNorm + SiLU ---------
__global__ void gat_aggregate_k(const unsigned* __restrict__ xh_u,
                                const float* __restrict__ a_src,
                                const float* __restrict__ a_dst,
                                const float* __restrict__ nlast,
                                const float* __restrict__ coef,   // [3][4] per layer
                                const float* __restrict__ gbias,  // [64]
                                const float* __restrict__ lng,
                                const float* __restrict__ lnb,
                                float* __restrict__ zout) {
  const int wid  = (blockIdx.x * blockDim.x + threadIdx.x) >> 5;
  const int lane = threadIdx.x & 31;
  const int b = wid >> 12, idx = wid & 4095, yy = idx >> 6, xx = idx & 63;

  // wave-uniform validity mask for the 8 incoming directions
  const unsigned up = yy > 0, dn = yy < GG - 1, lf = xx > 0, rt = xx < GG - 1;
  const unsigned vmask = ((up & lf) << 0) | (up << 1) | ((up & rt) << 2) |
                         (lf << 3) | (rt << 4) |
                         ((dn & lf) << 5) | (dn << 6) | ((dn & rt) << 7);

  // ---- phase 1: lanes = (neighbor, head); alpha / softmax across <=8 incoming edges ----
  const int nb = lane >> 2, h = lane & 3;
  bool valid = (vmask >> nb) & 1u;
  int src = (b << 12) + (yy - c_dy[nb]) * GG + (xx - c_dx[nb]);
  float alpha = -3.0e38f;
  if (valid) {
    float diff = nlast[wid] - nlast[src];
    float a = a_src[src * 4 + h] + a_dst[wid * 4 + h]
            + c_ndx[nb] * coef[h] + c_ndy[nb] * coef[4 + h] + diff * coef[8 + h];
    alpha = (a >= 0.f) ? a : 0.2f * a;                      // leaky relu
  }
  float amax = alpha;
#pragma unroll
  for (int s = 4; s < 32; s <<= 1) amax = fmaxf(amax, __shfl_xor(amax, s, 32));
  float ex = valid ? __expf(alpha - amax) : 0.f;
  float den = ex;
#pragma unroll
  for (int s = 4; s < 32; s <<= 1) den += __shfl_xor(den, s, 32);
  float wgt = ex / (den + 1e-16f);

  // ---- phase 2: lanes = feature pairs; weighted gather of neighbor features ----
  float acc0[HEADS] = {0.f, 0.f, 0.f, 0.f}, acc1[HEADS] = {0.f, 0.f, 0.f, 0.f};
  for (int n2 = 0; n2 < 8; ++n2) {
    if (!((vmask >> n2) & 1u)) continue;
    const unsigned* row = xh_u +
        (size_t)((b << 12) + (yy - c_dy[n2]) * GG + (xx - c_dx[n2])) * 128;
    __builtin_prefetch(row, 0, 0);                          // global_prefetch_b8 on gfx1250
#pragma unroll
    for (int hh = 0; hh < HEADS; ++hh) {
      float w = __shfl(wgt, n2 * 4 + hh, 32);
      unsigned pk = row[hh * 32 + lane];
      acc0[hh] += w * bf2f(pk & 0xffffu);
      acc1[hh] += w * bf2f(pk >> 16);
    }
  }
  const int c = 2 * lane;
  float v0 = 0.25f * (acc0[0] + acc0[1] + acc0[2] + acc0[3]) + gbias[c];
  float v1 = 0.25f * (acc1[0] + acc1[1] + acc1[2] + acc1[3]) + gbias[c + 1];
  // LayerNorm over 64 features (wave reduction) + SiLU
  float s1 = v0 + v1, s2 = v0 * v0 + v1 * v1;
#pragma unroll
  for (int s = 1; s < 32; s <<= 1) { s1 += __shfl_xor(s1, s, 32); s2 += __shfl_xor(s2, s, 32); }
  float mean = s1 * (1.f / 64.f);
  float var  = s2 * (1.f / 64.f) - mean * mean;
  float rstd = rsqrtf(var + 1e-5f);
  float z0 = (v0 - mean) * rstd * lng[c]     + lnb[c];
  float z1 = (v1 - mean) * rstd * lng[c + 1] + lnb[c + 1];
  zout[(size_t)wid * HID + c]     = silu_f(z0);
  zout[(size_t)wid * HID + c + 1] = silu_f(z1);
}

// ---------------- encoder: h = silu(nodes@w1+b1)@w2+b2 (wave per node) ----------------
__global__ void encoder_k(const float* __restrict__ x_in,
                          const float* __restrict__ w1, const float* __restrict__ b1,
                          const float* __restrict__ w2, const float* __restrict__ b2,
                          float* __restrict__ hout) {
  __shared__ float zsh[8][66];
  const int wid  = (blockIdx.x * blockDim.x + threadIdx.x) >> 5;
  const int wv   = threadIdx.x >> 5;
  const int lane = threadIdx.x & 31;
  const int b = wid >> 12, idx = wid & 4095;
  const float* xb = x_in + (size_t)b * 12 * NNODE + idx;
  float nd[12];
#pragma unroll
  for (int cc = 0; cc < 12; ++cc) nd[cc] = xb[(size_t)cc * NNODE];
  const int c = 2 * lane;
  float u0 = b1[c], u1 = b1[c + 1];
#pragma unroll
  for (int cc = 0; cc < 12; ++cc) {
    u0 += nd[cc] * w1[cc * 64 + c];
    u1 += nd[cc] * w1[cc * 64 + c + 1];
  }
  zsh[wv][c] = silu_f(u0); zsh[wv][c + 1] = silu_f(u1);
  __syncthreads();
  float o0 = b2[c], o1 = b2[c + 1];
  for (int k = 0; k < 64; ++k) {
    float zk = zsh[wv][k];
    o0 += zk * w2[k * 64 + c];
    o1 += zk * w2[k * 64 + c + 1];
  }
  hout[(size_t)wid * HID + c] = o0;
  hout[(size_t)wid * HID + c + 1] = o1;
}

// ---------------- head: LN -> silu(z@hw1+hb1)@hw2+hb2 -> fire mask ----------------
__global__ void head_k(const float* __restrict__ h,
                       const float* __restrict__ lg, const float* __restrict__ lb,
                       const float* __restrict__ hw1, const float* __restrict__ hb1,
                       const float* __restrict__ hw2, const float* __restrict__ hb2,
                       const float* __restrict__ x_in, float* __restrict__ out) {
  __shared__ float zsh[8][66];
  const int wid  = (blockIdx.x * blockDim.x + threadIdx.x) >> 5;
  const int wv   = threadIdx.x >> 5;
  const int lane = threadIdx.x & 31;
  const int c = 2 * lane;
  float v0 = h[(size_t)wid * HID + c], v1 = h[(size_t)wid * HID + c + 1];
  float s1 = v0 + v1, s2 = v0 * v0 + v1 * v1;
#pragma unroll
  for (int s = 1; s < 32; s <<= 1) { s1 += __shfl_xor(s1, s, 32); s2 += __shfl_xor(s2, s, 32); }
  float mean = s1 * (1.f / 64.f);
  float rstd = rsqrtf(s2 * (1.f / 64.f) - mean * mean + 1e-5f);
  zsh[wv][c]     = (v0 - mean) * rstd * lg[c]     + lb[c];
  zsh[wv][c + 1] = (v1 - mean) * rstd * lg[c + 1] + lb[c + 1];
  __syncthreads();
  float t0 = hb1[c], t1 = hb1[c + 1];
  for (int k = 0; k < 64; ++k) {
    float zk = zsh[wv][k];
    t0 += zk * hw1[k * 64 + c];
    t1 += zk * hw1[k * 64 + c + 1];
  }
  float part = silu_f(t0) * hw2[c] + silu_f(t1) * hw2[c + 1];
#pragma unroll
  for (int s = 16; s; s >>= 1) part += __shfl_xor(part, s, 32);
  if (lane == 0) {
    float logit = part + hb2[0];
    int b = wid >> 12, idx = wid & 4095;
    float fire = x_in[(size_t)b * 12 * NNODE + idx];     // channel 0
    out[wid] = (fire > 0.5f) ? fmaxf(logit, 6.0f) : logit;
  }
}

// ---------------- small helpers ----------------
__global__ void nlast_k(const float* __restrict__ x_in, float* __restrict__ nlast) {
  int i = blockIdx.x * blockDim.x + threadIdx.x;
  int b = i >> 12, idx = i & 4095;
  nlast[i] = x_in[(size_t)b * 12 * NNODE + (size_t)11 * NNODE + idx];
}
// a_e coefficients: coef[layer][e][h] = sum_c ew[e, h*64+c] * att_edge[h, c]
__global__ void precoef_k(const float* __restrict__ ew0, const float* __restrict__ ae0,
                          const float* __restrict__ ew1, const float* __restrict__ ae1,
                          float* __restrict__ coef) {
  int t = threadIdx.x;
  if (t < 24) {
    int layer = t / 12, r = t % 12, e = r >> 2, h = r & 3;
    const float* ew = layer ? ew1 : ew0;
    const float* ae = layer ? ae1 : ae0;
    float s = 0.f;
    for (int cc = 0; cc < 64; ++cc) s += ew[e * NCOL + h * 64 + cc] * ae[h * 64 + cc];
    coef[layer * 12 + e * 4 + h] = s;
  }
}
// RK4 elementwise stages, float4-vectorized (b128 traffic)
__global__ void rk_first_k(const float4* __restrict__ h, const float4* __restrict__ k,
                           float4* __restrict__ acc, float4* __restrict__ y,
                           float sacc, float sy) {
  int i = blockIdx.x * blockDim.x + threadIdx.x;
  float4 hv = h[i], kv = k[i];
  acc[i] = make_float4(hv.x + sacc * kv.x, hv.y + sacc * kv.y,
                       hv.z + sacc * kv.z, hv.w + sacc * kv.w);
  y[i]   = make_float4(hv.x + sy * kv.x, hv.y + sy * kv.y,
                       hv.z + sy * kv.z, hv.w + sy * kv.w);
}
__global__ void rk_mid_k(const float4* __restrict__ h, const float4* __restrict__ k,
                         float4* __restrict__ acc, float4* __restrict__ y,
                         float sacc, float sy) {
  int i = blockIdx.x * blockDim.x + threadIdx.x;
  float4 hv = h[i], kv = k[i], av = acc[i];
  acc[i] = make_float4(av.x + sacc * kv.x, av.y + sacc * kv.y,
                       av.z + sacc * kv.z, av.w + sacc * kv.w);
  y[i]   = make_float4(hv.x + sy * kv.x, hv.y + sy * kv.y,
                       hv.z + sy * kv.z, hv.w + sy * kv.w);
}
__global__ void rk_last_k(const float4* __restrict__ acc, const float4* __restrict__ k,
                          float4* __restrict__ h, float sacc) {
  int i = blockIdx.x * blockDim.x + threadIdx.x;
  float4 av = acc[i], kv = k[i];
  h[i] = make_float4(av.x + sacc * kv.x, av.y + sacc * kv.y,
                     av.z + sacc * kv.z, av.w + sacc * kv.w);
}

// ---------------- launcher ----------------
extern "C" void kernel_launch(void* const* d_in, const int* in_sizes, int n_in,
                              void* d_out, int out_size, void* d_ws, size_t ws_size,
                              hipStream_t stream) {
  (void)in_sizes; (void)n_in; (void)out_size; (void)ws_size;
  const float* x      = (const float*)d_in[0];
  const float* enc_w1 = (const float*)d_in[1];
  const float* enc_b1 = (const float*)d_in[2];
  const float* enc_w2 = (const float*)d_in[3];
  const float* enc_b2 = (const float*)d_in[4];
  const float* g_w [2] = {(const float*)d_in[5],  (const float*)d_in[11]};
  const float* g_ew[2] = {(const float*)d_in[6],  (const float*)d_in[12]};
  const float* g_as[2] = {(const float*)d_in[7],  (const float*)d_in[13]};
  const float* g_ad[2] = {(const float*)d_in[8],  (const float*)d_in[14]};
  const float* g_ae[2] = {(const float*)d_in[9],  (const float*)d_in[15]};
  const float* g_b [2] = {(const float*)d_in[10], (const float*)d_in[16]};
  const float* n_g [2] = {(const float*)d_in[17], (const float*)d_in[19]};
  const float* n_b [2] = {(const float*)d_in[18], (const float*)d_in[20]};
  const float* hd_lg = (const float*)d_in[21];
  const float* hd_lb = (const float*)d_in[22];
  const float* hd_w1 = (const float*)d_in[23];
  const float* hd_b1 = (const float*)d_in[24];
  const float* hd_w2 = (const float*)d_in[25];
  const float* hd_b2 = (const float*)d_in[26];
  // d_in[27]=edge_index, d_in[28]=edge_dirs: unused (static 8-neighbor grid exploited)

  char* p = (char*)d_ws;
  const size_t SZH = (size_t)MTOT * HID * sizeof(float);
  float* hbuf = (float*)p; p += SZH;
  float* ybuf = (float*)p; p += SZH;
  float* abuf = (float*)p; p += SZH;
  float* kbuf = (float*)p; p += SZH;
  float* zbuf = (float*)p; p += SZH;
  unsigned short* xh = (unsigned short*)p; p += (size_t)MTOT * NCOL * sizeof(unsigned short);
  float* asrc  = (float*)p; p += (size_t)MTOT * HEADS * sizeof(float);
  float* adst  = (float*)p; p += (size_t)MTOT * HEADS * sizeof(float);
  float* nlast = (float*)p; p += (size_t)MTOT * sizeof(float);
  float* coef  = (float*)p; p += 256;
  unsigned short* wbf = (unsigned short*)p; p += (size_t)2 * NCOL * HID * sizeof(unsigned short);

  nlast_k  <<<MTOT / 256, 256, 0, stream>>>(x, nlast);
  precoef_k<<<1, 32, 0, stream>>>(g_ew[0], g_ae[0], g_ew[1], g_ae[1], coef);
  convw_k  <<<(2 * NCOL * HID) / 256, 256, 0, stream>>>(g_w[0], g_w[1], wbf);
  encoder_k<<<MTOT / 8, 256, 0, stream>>>(x, enc_w1, enc_b1, enc_w2, enc_b2, hbuf);

  auto runF = [&](const float* hin, float* kout) {
    for (int L = 0; L < 2; ++L) {
      const float* zin = (L == 0) ? hin : zbuf;
      float*       zo  = (L == 0) ? zbuf : kout;
      wmma_gemm64     <<<MTOT / 128, 256, 0, stream>>>(zin, wbf + (size_t)L * NCOL * HID,
                                                       g_as[L], g_ad[L], xh, asrc, adst);
      gat_aggregate_k <<<MTOT / 8,   256, 0, stream>>>((const unsigned*)xh, asrc, adst, nlast,
                                                       coef + 12 * L, g_b[L], n_g[L], n_b[L], zo);
    }
  };

  const float dt = 0.5f;                       // T_END / N_EVAL
  const int NB4 = (MTOT * HID / 4) / 256;      // float4 grid
  for (int step = 0; step < 2; ++step) {
    runF(hbuf, kbuf);
    rk_first_k<<<NB4, 256, 0, stream>>>((const float4*)hbuf, (const float4*)kbuf,
                                        (float4*)abuf, (float4*)ybuf, dt / 6.f, dt * 0.5f);
    runF(ybuf, kbuf);
    rk_mid_k  <<<NB4, 256, 0, stream>>>((const float4*)hbuf, (const float4*)kbuf,
                                        (float4*)abuf, (float4*)ybuf, dt / 3.f, dt * 0.5f);
    runF(ybuf, kbuf);
    rk_mid_k  <<<NB4, 256, 0, stream>>>((const float4*)hbuf, (const float4*)kbuf,
                                        (float4*)abuf, (float4*)ybuf, dt / 3.f, dt);
    runF(ybuf, kbuf);
    rk_last_k <<<NB4, 256, 0, stream>>>((const float4*)abuf, (const float4*)kbuf,
                                        (float4*)hbuf, dt / 6.f);
  }
  head_k<<<MTOT / 8, 256, 0, stream>>>(hbuf, hd_lg, hd_lb, hd_w1, hd_b1, hd_w2, hd_b2,
                                       x, (float*)d_out);
}
